// FDPKT_34471407518130
// MI455X (gfx1250) — compile-verified, matrix operand order checked by reading, not run on hardware
//
#include <hip/hip_runtime.h>

#define B_N 1024
#define S_N 199
#define D_N 128
#define PRO_N 10000
#define ALGO_N 200

typedef __attribute__((ext_vector_type(16))) __bf16 v16bf;
typedef __attribute__((ext_vector_type(8)))  float  v8f;

static __device__ __forceinline__ unsigned short f2bs(float f) {
  return __builtin_bit_cast(unsigned short, (__bf16)f);
}
static __device__ __forceinline__ float bs2f(unsigned short u) {
  unsigned int x = ((unsigned int)u) << 16;
  return __builtin_bit_cast(float, x);
}
static __device__ __forceinline__ float sigmf(float x) {
  return 1.0f / (1.0f + __expf(-x));
}

// ---------------------------------------------------------------------------
// Prep kernel 1: per-(b,s) embedding + response rows, stored bf16 in workspace
//   proEmb = pro_embed[p] + algo_embed[a] + akt_pro_diff[p]*akt_pro_change[a]
//   R      = (ans>=0.5 ? ps : 1-ps) * err_embed[e]
// ---------------------------------------------------------------------------
__global__ __launch_bounds__(128) void k_embed(
    const int* __restrict__ nprob, const int* __restrict__ nalgo,
    const int* __restrict__ nerr, const float* __restrict__ nans,
    const float* __restrict__ nps, const float* __restrict__ proE,
    const float* __restrict__ algoE, const float* __restrict__ errE,
    const float* __restrict__ diff, const float* __restrict__ chg,
    unsigned short* __restrict__ gPro, unsigned short* __restrict__ gR) {
  const int bs = blockIdx.x;   // b*S + s
  const int d  = threadIdx.x;  // 0..127
  const int p = nprob[bs], a = nalgo[bs], e = nerr[bs];
  const float w = (nans[bs] >= 0.5f) ? nps[bs] : 1.0f - nps[bs];
  const float pe = proE[p * D_N + d] + algoE[a * D_N + d] + diff[p] * chg[a * D_N + d];
  const float r  = w * errE[e * D_N + d];
  gPro[(size_t)bs * D_N + d] = f2bs(pe);
  gR[(size_t)bs * D_N + d]   = f2bs(r);
}

// ---------------------------------------------------------------------------
// Prep kernel 2: swizzle the 7 weight matrices [K x 128] f32 into the per-lane
// WMMA B-fragment order (bf16): [ntile][ktile][lane][16 elems] contiguous.
// K-mapping matches the documented 16-bit operand layout:
//   elem e (0..15), i=e>>1: K = kt*32 + (lane>=16 ? 8:0) + (i&4?16:0) + 2*(i&3) + (e&1)
// ---------------------------------------------------------------------------
struct SwArgs { const float* w[7]; };

__global__ __launch_bounds__(256) void k_swz(SwArgs A, unsigned short* __restrict__ wsw) {
  const int cum[7]      = {64, 128, 224, 288, 384, 480, 576};
  const int kt_[7]      = {8, 8, 12, 8, 12, 12, 12};
  const unsigned off_[7]= {0u, 32768u, 65536u, 114688u, 147456u, 196608u, 245760u};
  int bid = blockIdx.x;  // 0..575
  int g = 0;
  while (bid >= cum[g]) ++g;
  const int tix = bid - (g ? cum[g - 1] : 0);
  const int Kt  = kt_[g];
  const int ntl = tix / Kt;
  const int kt  = tix % Kt;
  for (int e2 = threadIdx.x; e2 < 512; e2 += 256) {
    const int ln = e2 >> 4, e = e2 & 15;
    const int i  = e >> 1;
    const int k  = kt * 32 + ((i & 4) ? 16 : 0) + ((ln >> 4) << 3) + ((i & 3) << 1) + (e & 1);
    const int n  = ntl * 16 + (ln & 15);
    const float v = A.w[g][(size_t)k * D_N + n];
    wsw[off_[g] + ((size_t)(ntl * Kt + kt) * 32 + ln) * 16 + e] = f2bs(v);
  }
}

// ---------------------------------------------------------------------------
// WMMA 16xK x Kx16 tile GEMM for one wave (ntile = wave id). A comes from LDS
// row-major [16][384] bf16 (K-pairs are dword-contiguous), B is pre-swizzled.
// ---------------------------------------------------------------------------
union ARegs { v16bf v; unsigned int d[8]; };

static __device__ __forceinline__ v8f
gemm16(const unsigned short* __restrict__ sA, int ktiles,
       const unsigned short* __restrict__ wsw, const float* __restrict__ bias,
       int lane, int nt) {
  const float bv = bias[nt * 16 + (lane & 15)];
  v8f c = {bv, bv, bv, bv, bv, bv, bv, bv};
  const unsigned short* arow = sA + (size_t)(lane & 15) * 384;
  const int kb = (lane >> 4) << 3;
  for (int kt = 0; kt < ktiles; ++kt) {
    ARegs a;
    const int kk = kt * 32 + kb;
#pragma unroll
    for (int i = 0; i < 8; ++i) {
      const int k = kk + ((i & 4) ? 16 : 0) + ((i & 3) << 1);
      a.d[i] = *(const unsigned int*)(arow + k);
    }
    const v16bf b = *(const v16bf*)(wsw + ((size_t)(nt * ktiles + kt) * 32 + lane) * 16);
    c = __builtin_amdgcn_wmma_f32_16x16x32_bf16(false, a.v, false, b, (short)0, c,
                                                false, false);
  }
  return c;
}

// ---------------------------------------------------------------------------
// Persistent scan kernel: 64 WGs x 256 threads (8 waves). Each WG owns 16
// samples and runs all 199 steps. All recurrent tables live in LDS (247 KB).
// ---------------------------------------------------------------------------
struct ScanArgs {
  const int* nprob;
  const int* nalgo;
  const float* time_embed;
  const float* ls_state;
  const float* algo_state0;
  const float* out2_w;
  const float* out2_b;
  const float* bias[7];  // algo_forget, cl_forget, out1, router, change, cl_obtain, algo_obtain
  const unsigned short* wsw;
  const unsigned short* gPro;
  const unsigned short* gR;
  float* gAlgo;  // [B][S][D] f32 history
  float* out;    // [B][S]
};

#define WSW_AF 0u
#define WSW_CF 32768u
#define WSW_O1 65536u
#define WSW_RO 114688u
#define WSW_CH 147456u
#define WSW_CO 196608u
#define WSW_AO 245760u

__global__ __launch_bounds__(256) void k_scan(ScanArgs A) {
  extern __shared__ char smem[];
  float* sLastA = (float*)smem;            // 16*128 f32
  float* sCl    = sLastA + 2048;           // 16*128 f32 (persistent cl_state)
  float* sLbcl  = sCl + 2048;              // 16*128
  float* sGate  = sLbcl + 2048;            // 16*128
  float* sH     = sGate + 2048;            // 16*128
  int* sLat     = (int*)(sH + 2048);       // 16*200 last_algo_time
  int* sScal    = sLat + 16 * ALGO_N;      // 16*4  (p, a, lat, lpt)
  unsigned short* sA   = (unsigned short*)(sScal + 64);  // 16*384 bf16 staging
  unsigned short* sPro = sA + 16 * 384;    // 16*128 bf16
  unsigned short* sR   = sPro + 2048;
  unsigned short* sX   = sR + 2048;
  unsigned short* sLpr = sX + 2048;
  unsigned short* sChg = sLpr + 2048;
  unsigned short* sCgap = sChg + 2048;     // 128 bf16 (time_embed[1])
  unsigned char* sLpt  = (unsigned char*)(sCgap + 128);  // 16*10000 (t+1, 0=none)

  const int tid  = threadIdx.x;
  const int lane = tid & 31;
  const int nt   = tid >> 5;       // wave id == ntile 0..7
  const int b0   = blockIdx.x * 16;
  const int s    = tid >> 4;       // staging sample 0..15
  const int c0   = (tid & 15) * 8; // 8 columns per staging thread
  const int col  = nt * 16 + (lane & 15);
  const int rb   = (lane >> 4) << 3;

  // ---- init persistent LDS state ----
  for (int i = tid; i < 16 * ALGO_N; i += 256) sLat[i] = 0;
  for (int i = tid * 4; i < 16 * PRO_N; i += 256 * 4)
    *(unsigned int*)(sLpt + i) = 0u;
#pragma unroll
  for (int j = 0; j < 8; ++j) sCl[s * D_N + c0 + j] = A.ls_state[c0 + j];
  if (tid < 16) {
#pragma unroll
    for (int j = 0; j < 8; ++j)
      sCgap[tid * 8 + j] = f2bs(A.time_embed[D_N + tid * 8 + j]);
  }
  __syncthreads();

  for (int t = 0; t < S_N; ++t) {
    // ---- phase A: per-sample scalars ----
    if (tid < 16) {
      const int p = A.nprob[(size_t)(b0 + tid) * S_N + t];
      const int a = A.nalgo[(size_t)(b0 + tid) * S_N + t];
      sScal[tid * 4 + 0] = p;
      sScal[tid * 4 + 1] = a;
      sScal[tid * 4 + 2] = sLat[tid * ALGO_N + a];
      sScal[tid * 4 + 3] = (int)sLpt[tid * PRO_N + p] - 1;
    }
    __syncthreads();
    const int lat = sScal[s * 4 + 2];
    const int lpt = sScal[s * 4 + 3];

    // ---- phase B: gather + stage A1 = [last_a | gap_emb] ----
    {
      const float* laSrc = (t == 0)
          ? A.algo_state0
          : (A.gAlgo + ((size_t)(b0 + s) * S_N + lat) * D_N);
      const float* gapSrc = A.time_embed + (size_t)(t - lat) * D_N;
      const unsigned short* peSrc = A.gPro + ((size_t)(b0 + s) * S_N + t) * D_N;
      const unsigned short* rSrc  = A.gR + ((size_t)(b0 + s) * S_N + t) * D_N;
      const unsigned short* lprSrc =
          (lpt >= 0) ? (A.gR + ((size_t)(b0 + s) * S_N + lpt) * D_N) : nullptr;
#pragma unroll
      for (int j = 0; j < 8; ++j) {
        const int cc = c0 + j;
        const float la = laSrc[cc];
        sLastA[s * D_N + cc]  = la;
        sA[s * 384 + cc]       = f2bs(la);
        sA[s * 384 + 128 + cc] = f2bs(gapSrc[cc]);
        const unsigned short pe = peSrc[cc], rr = rSrc[cc];
        sPro[s * D_N + cc] = pe;
        sR[s * D_N + cc]   = rr;
        sX[s * D_N + cc]   = f2bs(bs2f(pe) + bs2f(rr));
        sLpr[s * D_N + cc] = lprSrc ? lprSrc[cc] : (unsigned short)0;
      }
    }
    __syncthreads();

    // ---- GEMM1: algo_forget (K=256); last_a *= sigmoid ----
    {
      v8f c = gemm16(sA, 8, A.wsw + WSW_AF, A.bias[0], lane, nt);
#pragma unroll
      for (int r = 0; r < 8; ++r) sLastA[(rb + r) * D_N + col] *= sigmf(c[r]);
    }
    __syncthreads();

    // ---- stage A2 = [cl_state | cl_gap] ----
#pragma unroll
    for (int j = 0; j < 8; ++j) {
      const int cc = c0 + j;
      sA[s * 384 + cc]       = f2bs(sCl[s * D_N + cc]);
      sA[s * 384 + 128 + cc] = sCgap[cc];
    }
    __syncthreads();

    // ---- GEMM2: cl_forget; lbcl = cl * sigmoid ----
    {
      v8f c = gemm16(sA, 8, A.wsw + WSW_CF, A.bias[1], lane, nt);
#pragma unroll
      for (int r = 0; r < 8; ++r) {
        const int idx = (rb + r) * D_N + col;
        sLbcl[idx] = sCl[idx] * sigmf(c[r]);
      }
    }
    __syncthreads();

    // ---- stage A3 = [lbcl | last_a | pro_emb] ----
#pragma unroll
    for (int j = 0; j < 8; ++j) {
      const int cc = c0 + j;
      sA[s * 384 + cc]       = f2bs(sLbcl[s * D_N + cc]);
      sA[s * 384 + 128 + cc] = f2bs(sLastA[s * D_N + cc]);
      sA[s * 384 + 256 + cc] = sPro[s * D_N + cc];
    }
    __syncthreads();

    // ---- GEMM3: out1 (K=384); h = relu ----
    {
      v8f c = gemm16(sA, 12, A.wsw + WSW_O1, A.bias[2], lane, nt);
#pragma unroll
      for (int r = 0; r < 8; ++r) sH[(rb + r) * D_N + col] = fmaxf(c[r], 0.0f);
    }
    __syncthreads();

    // ---- stage A4 = [pro_emb | R]; out2 head: P = sigmoid(h . w2 + b2) ----
#pragma unroll
    for (int j = 0; j < 8; ++j) {
      const int cc = c0 + j;
      sA[s * 384 + cc]       = sPro[s * D_N + cc];
      sA[s * 384 + 128 + cc] = sR[s * D_N + cc];
    }
    if (tid < 16) {
      float acc = A.out2_b[0];
      for (int i = 0; i < D_N; ++i) acc += sH[tid * D_N + i] * A.out2_w[i];
      A.out[(size_t)(b0 + tid) * S_N + t] = sigmf(acc);
    }
    __syncthreads();

    // ---- GEMM4: router; gate = sigmoid ----
    {
      v8f c = gemm16(sA, 8, A.wsw + WSW_RO, A.bias[3], lane, nt);
#pragma unroll
      for (int r = 0; r < 8; ++r) sGate[(rb + r) * D_N + col] = sigmf(c[r]);
    }
    __syncthreads();

    // ---- stage A5 = [R | R - lastProR | R * lastProR] ----
#pragma unroll
    for (int j = 0; j < 8; ++j) {
      const int cc = c0 + j;
      const float r = bs2f(sR[s * D_N + cc]);
      const float q = bs2f(sLpr[s * D_N + cc]);
      sA[s * 384 + cc]       = sR[s * D_N + cc];
      sA[s * 384 + 128 + cc] = f2bs(r - q);
      sA[s * 384 + 256 + cc] = f2bs(r * q);
    }
    __syncthreads();

    // ---- GEMM5: change (K=384, linear) -> bf16 ----
    {
      v8f c = gemm16(sA, 12, A.wsw + WSW_CH, A.bias[4], lane, nt);
#pragma unroll
      for (int r = 0; r < 8; ++r) sChg[(rb + r) * D_N + col] = f2bs(c[r]);
    }
    __syncthreads();

    // ---- stage A6 = [lbcl | X | change] ----
#pragma unroll
    for (int j = 0; j < 8; ++j) {
      const int cc = c0 + j;
      sA[s * 384 + cc]       = f2bs(sLbcl[s * D_N + cc]);
      sA[s * 384 + 128 + cc] = sX[s * D_N + cc];
      sA[s * 384 + 256 + cc] = sChg[s * D_N + cc];
    }
    __syncthreads();

    // ---- GEMM6: cl_obtain; cl = lbcl + gate * tanh ----
    {
      v8f c = gemm16(sA, 12, A.wsw + WSW_CO, A.bias[5], lane, nt);
#pragma unroll
      for (int r = 0; r < 8; ++r) {
        const int idx = (rb + r) * D_N + col;
        sCl[idx] = sLbcl[idx] + sGate[idx] * tanhf(c[r]);
      }
    }
    __syncthreads();

    // ---- stage A7 = [last_a | X | change] ----
#pragma unroll
    for (int j = 0; j < 8; ++j) {
      const int cc = c0 + j;
      sA[s * 384 + cc]       = f2bs(sLastA[s * D_N + cc]);
      sA[s * 384 + 128 + cc] = sX[s * D_N + cc];
      sA[s * 384 + 256 + cc] = sChg[s * D_N + cc];
    }
    __syncthreads();

    // ---- GEMM7: algo_obtain; a_new = last_a + (1-gate)*tanh -> HBM history ----
    {
      v8f c = gemm16(sA, 12, A.wsw + WSW_AO, A.bias[6], lane, nt);
#pragma unroll
      for (int r = 0; r < 8; ++r) {
        const int row = rb + r;
        const int idx = row * D_N + col;
        const float an = sLastA[idx] + (1.0f - sGate[idx]) * tanhf(c[r]);
        A.gAlgo[((size_t)(b0 + row) * S_N + t) * D_N + col] = an;
      }
    }
    // ---- scatter visit-time updates ----
    if (tid < 16) {
      sLat[tid * ALGO_N + sScal[tid * 4 + 1]] = t;
      sLpt[tid * PRO_N + sScal[tid * 4 + 0]] = (unsigned char)(t + 1);
    }
    __threadfence();  // make this step's gAlgo row visible to next-step gathers
    __syncthreads();
  }
}

// ---------------------------------------------------------------------------
// Workspace layout (bytes):
//   [0)              wsw      bf16 swizzled weights  589,824
//   [589824)         gPro     bf16 B*S*D             52,166,656
//   [+)              gR       bf16 B*S*D             52,166,656
//   [+)              gAlgo    f32  B*S*D             104,333,312
//   total ~209.3 MB
// ---------------------------------------------------------------------------
extern "C" void kernel_launch(void* const* d_in, const int* in_sizes, int n_in,
                              void* d_out, int out_size, void* d_ws, size_t ws_size,
                              hipStream_t stream) {
  (void)in_sizes; (void)n_in; (void)out_size; (void)ws_size;
  const int*   nprob  = (const int*)d_in[3];
  const int*   nalgo  = (const int*)d_in[4];
  const float* nans   = (const float*)d_in[5];
  const int*   nerr   = (const int*)d_in[7];
  const float* nps    = (const float*)d_in[9];
  const float* proE   = (const float*)d_in[10];
  const float* algoE  = (const float*)d_in[11];
  const float* errE   = (const float*)d_in[12];
  const float* timeE  = (const float*)d_in[13];
  const float* lsSt   = (const float*)d_in[14];
  const float* aSt0   = (const float*)d_in[15];
  const float* diff   = (const float*)d_in[16];
  const float* chg    = (const float*)d_in[17];
  const float* out1w  = (const float*)d_in[18];
  const float* out1b  = (const float*)d_in[19];
  const float* out2w  = (const float*)d_in[20];
  const float* out2b  = (const float*)d_in[21];
  const float* afW = (const float*)d_in[22]; const float* afB = (const float*)d_in[23];
  const float* cfW = (const float*)d_in[24]; const float* cfB = (const float*)d_in[25];
  const float* roW = (const float*)d_in[26]; const float* roB = (const float*)d_in[27];
  const float* chW = (const float*)d_in[28]; const float* chB = (const float*)d_in[29];
  const float* coW = (const float*)d_in[30]; const float* coB = (const float*)d_in[31];
  const float* aoW = (const float*)d_in[32]; const float* aoB = (const float*)d_in[33];

  char* ws = (char*)d_ws;
  unsigned short* wsw  = (unsigned short*)ws;
  unsigned short* gPro = (unsigned short*)(ws + 589824);
  unsigned short* gR   = (unsigned short*)(ws + 589824 + 52166656ull);
  float*          gAlg = (float*)(ws + 589824 + 2ull * 52166656ull);

  k_embed<<<B_N * S_N, 128, 0, stream>>>(nprob, nalgo, nerr, nans, nps, proE,
                                         algoE, errE, diff, chg, gPro, gR);

  SwArgs sw;
  sw.w[0] = afW; sw.w[1] = cfW; sw.w[2] = out1w; sw.w[3] = roW;
  sw.w[4] = chW; sw.w[5] = coW; sw.w[6] = aoW;
  k_swz<<<576, 256, 0, stream>>>(sw, wsw);

  ScanArgs A;
  A.nprob = nprob; A.nalgo = nalgo; A.time_embed = timeE; A.ls_state = lsSt;
  A.algo_state0 = aSt0; A.out2_w = out2w; A.out2_b = out2b;
  A.bias[0] = afB; A.bias[1] = cfB; A.bias[2] = out1b; A.bias[3] = roB;
  A.bias[4] = chB; A.bias[5] = coB; A.bias[6] = aoB;
  A.wsw = wsw; A.gPro = gPro; A.gR = gR; A.gAlgo = gAlg; A.out = (float*)d_out;

  // Dynamic LDS: 5 f32 bufs + tables + bf16 staging + u8 lpt = 247,040 B
  constexpr size_t SMEM = 5 * 2048 * 4 + (16 * ALGO_N + 64) * 4 +
                          (16 * 384 + 5 * 2048 + 128) * 2 + 16 * PRO_N;
  k_scan<<<64, 256, SMEM, stream>>>(A);
}
